// CNN_SoftDTW_37890201485905
// MI455X (gfx1250) — compile-verified, matrix operand order checked by reading, not run on hardware
//
#include <hip/hip_runtime.h>
#include <hip/hip_bf16.h>

typedef _Float16 v16h __attribute__((ext_vector_type(16)));
typedef _Float16 h8   __attribute__((ext_vector_type(8)));
typedef float    v8f  __attribute__((ext_vector_type(8)));

#define TT   128      // time steps (fixed by reference)
#define KC   32       // K-chunk per WMMA (f16 16x16x32)
#define KP   40       // padded LDS row (halves): 80B -> 16B aligned, spread banks
#define BIGF 1e10f
#define GAM  0.1f

__device__ __forceinline__ float softmin3(float a, float b, float c) {
    // -g * logsumexp(-s/g) = m - g*log(sum exp((m - s_k)/g)), m = min
    float m  = fminf(a, fminf(b, c));
    float e  = __expf((m - a) * (1.0f / GAM)) +
               __expf((m - b) * (1.0f / GAM)) +
               __expf((m - c) * (1.0f / GAM));
    return m - GAM * __logf(e);
}

// ---------------------------------------------------------------------------
// Kernel 1: squared row norms for target (Q*T rows) and support (S*T rows)
// ---------------------------------------------------------------------------
__global__ void row_sqnorm_kernel(const float* __restrict__ Xg,
                                  const float* __restrict__ Yg,
                                  float* __restrict__ x2, float* __restrict__ y2,
                                  int nX, int nY, int D)
{
    int row  = blockIdx.x * 8 + (threadIdx.x >> 5);
    int lane = threadIdx.x & 31;
    if (row >= nX + nY) return;
    const float* p = (row < nX) ? (Xg + (size_t)row * D)
                                : (Yg + (size_t)(row - nX) * D);
    float sum = 0.0f;
    for (int k = lane; k < D; k += 32) { float v = p[k]; sum += v * v; }
#pragma unroll
    for (int off = 16; off > 0; off >>= 1) sum += __shfl_xor(sum, off, 32);
    if (lane == 0) {
        if (row < nX) x2[row] = sum; else y2[row - nX] = sum;
    }
}

// ---------------------------------------------------------------------------
// Kernel 2: fused (pairwise sq-dist via WMMA f16 GEMM) + soft-DTW per (q,s)
// One workgroup = one (q,s) pair. 256 threads = 8 wave32.
// Double-buffered LDS staging (one barrier per K-chunk); global loads for
// chunk c+1 are issued before the WMMAs of chunk c. All 8 B fragments are
// preloaded into distinct registers so the 8 WMMAs issue back-to-back with
// staggered s_wait_dscnt instead of a full LDS round-trip stall per WMMA.
// ---------------------------------------------------------------------------
__global__ __launch_bounds__(256, 1)
void sdtw_pair_kernel(const float* __restrict__ Xg,   // [Q,T,D] target
                      const float* __restrict__ Yg,   // [S,T,D] support
                      const float* __restrict__ x2g,  // [Q*T]
                      const float* __restrict__ y2g,  // [S*T]
                      float* __restrict__ cum,        // [Q*S]
                      int S, int D)
{
    __shared__ __align__(16) _Float16 xT[2][TT][KP];  // 20 KB
    __shared__ __align__(16) _Float16 yT[2][TT][KP];  // 20 KB
    __shared__ float Dc[TT][TT];                      // 64 KB cost tile
    __shared__ float x2r[TT], y2r[TT];
    __shared__ float dgA[TT + 2], dgB[TT + 2];        // rotating anti-diagonals

    const int q    = blockIdx.x / S;
    const int s    = blockIdx.x - q * S;
    const int tid  = threadIdx.x;
    const int wave = tid >> 5;
    const int lane = tid & 31;

    if (tid < TT) { x2r[tid] = x2g[q * TT + tid]; y2r[tid] = y2g[s * TT + tid]; }

    const float* Xq = Xg + (size_t)q * TT * D;
    const float* Ys = Yg + (size_t)s * TT * D;

    v8f acc[8] = {};                 // 8 column tiles of 16x16 f32 accum

    // staging: 2 threads per row, each moves 16 f32 -> f16 per chunk
    const int m    = tid >> 1;
    const int h16  = (tid & 1) * 16;
    const float* xRow = Xq + (size_t)m * D + h16;
    const float* yRow = Ys + (size_t)m * D + h16;
    // WMMA 16-bit A/B fragment addressing (ISA 7.12.2):
    // lane&15 = M (resp. N), (lane>>4)*8 = K half-block base
    const int am = 16 * wave + (lane & 15);
    const int kb = (lane >> 4) * 8;

    union Frag { v16h v; h8 h[2]; };

    float4 fx[4], fy[4];

    // ---- prologue: stage chunk 0 into buffer 0 ----
#pragma unroll
    for (int v = 0; v < 4; ++v) {
        fx[v] = *(const float4*)(xRow + v * 4);
        fy[v] = *(const float4*)(yRow + v * 4);
    }
#pragma unroll
    for (int v = 0; v < 4; ++v) {
        int c0 = h16 + v * 4;
        xT[0][m][c0 + 0] = (_Float16)fx[v].x;  xT[0][m][c0 + 1] = (_Float16)fx[v].y;
        xT[0][m][c0 + 2] = (_Float16)fx[v].z;  xT[0][m][c0 + 3] = (_Float16)fx[v].w;
        yT[0][m][c0 + 0] = (_Float16)fy[v].x;  yT[0][m][c0 + 1] = (_Float16)fy[v].y;
        yT[0][m][c0 + 2] = (_Float16)fy[v].z;  yT[0][m][c0 + 3] = (_Float16)fy[v].w;
    }
    __syncthreads();

    const int nc = D / KC;
    int p = 0;
    for (int c = 0; c < nc; ++c) {
        // (a) issue global loads for chunk c+1 (latency hidden behind WMMAs)
        if (c + 1 < nc) {
            const float* xn = xRow + (size_t)(c + 1) * KC;
            const float* yn = yRow + (size_t)(c + 1) * KC;
#pragma unroll
            for (int v = 0; v < 4; ++v) {
                fx[v] = *(const float4*)(xn + v * 4);
                fy[v] = *(const float4*)(yn + v * 4);
            }
        }

        // (b) preload A + all 8 B fragments, then 8 back-to-back WMMAs
        const _Float16 (*xb)[KP] = xT[p];
        const _Float16 (*yb)[KP] = yT[p];
        Frag af, bf[8];
        af.h[0] = *(const h8*)&xb[am][kb];
        af.h[1] = *(const h8*)&xb[am][kb + 16];
#pragma unroll
        for (int nt = 0; nt < 8; ++nt) {
            const int bn = nt * 16 + (lane & 15);
            bf[nt].h[0] = *(const h8*)&yb[bn][kb];
            bf[nt].h[1] = *(const h8*)&yb[bn][kb + 16];
        }
#pragma unroll
        for (int nt = 0; nt < 8; ++nt) {
            acc[nt] = __builtin_amdgcn_wmma_f32_16x16x32_f16(
                false, af.v, false, bf[nt].v, (short)0, acc[nt], false, false);
        }

        // (c) convert + store chunk c+1 into the other buffer
        if (c + 1 < nc) {
            const int pn = p ^ 1;
#pragma unroll
            for (int v = 0; v < 4; ++v) {
                int c0 = h16 + v * 4;
                xT[pn][m][c0 + 0] = (_Float16)fx[v].x;  xT[pn][m][c0 + 1] = (_Float16)fx[v].y;
                xT[pn][m][c0 + 2] = (_Float16)fx[v].z;  xT[pn][m][c0 + 3] = (_Float16)fx[v].w;
                yT[pn][m][c0 + 0] = (_Float16)fy[v].x;  yT[pn][m][c0 + 1] = (_Float16)fy[v].y;
                yT[pn][m][c0 + 2] = (_Float16)fy[v].z;  yT[pn][m][c0 + 3] = (_Float16)fy[v].w;
            }
        }

        // (d) single barrier per chunk: separates this chunk's frag reads /
        //     next-buffer stores from the next chunk's reads.
        __syncthreads();
        p ^= 1;
    }

    // Cost tile: D[i][j] = |x_i|^2 + |y_j|^2 - 2 * <x_i, y_j>
    // C/D layout: VGPR r -> M = r + 8*(lane>>4), lanes&15 -> N
    {
        const int mb = 16 * wave + 8 * (lane >> 4);
        const int nb = lane & 15;
#pragma unroll
        for (int nt = 0; nt < 8; ++nt)
#pragma unroll
            for (int r = 0; r < 8; ++r) {
                int M = mb + r;
                int N = nt * 16 + nb;
                Dc[M][N] = x2r[M] + y2r[N] - 2.0f * acc[nt][r];
            }
    }

    // ---- soft-DTW anti-diagonal wavefront ----
    if (tid < TT + 2) { dgA[tid] = BIGF; dgB[tid] = BIGF; }
    if (tid == 0) dgA[0] = 0.0f;      // R[-1,-1] = 0
    __syncthreads();

    float* A = dgA;   // diag d-2 (also receives diag d)
    float* B = dgB;   // diag d-1
    for (int d = 0; d < 2 * TT - 1; ++d) {
        float u = 0.f, l = 0.f, g = 0.f, dc = 0.f;
        bool act = false;
        const int j = tid;
        if (tid < TT) {
            int i = d - j;
            if (i >= 0 && i < TT) {
                act = true;
                u  = B[j + 1];   // R[i-1, j]
                l  = B[j];       // R[i,   j-1]
                g  = A[j];       // R[i-1, j-1]
                dc = Dc[i][j];
            }
        }
        __syncthreads();
        if (act) A[j + 1] = dc + softmin3(u, l, g);
        if (tid == 0) A[0] = BIGF;   // left border stays +inf
        __syncthreads();
        float* t = A; A = B; B = t;
    }
    if (tid == 0) cum[blockIdx.x] = B[TT];   // R[T,T]
}

// ---------------------------------------------------------------------------
// Kernel 3: per-class mean + negate. Labels are int64.
// ---------------------------------------------------------------------------
__global__ void aggregate_kernel(const float* __restrict__ cum,
                                 const long long* __restrict__ labels,
                                 float* __restrict__ out, int Q, int S, int C)
{
    int q = blockIdx.x * blockDim.x + threadIdx.x;
    if (q >= Q) return;
    for (int c = 0; c < C; ++c) {
        float sum = 0.0f, cnt = 0.0f;
        for (int s = 0; s < S; ++s) {
            if ((int)labels[s] == c) { sum += cum[q * S + s]; cnt += 1.0f; }
        }
        out[q * C + c] = -(sum / fmaxf(cnt, 1.0f));
    }
}

// ---------------------------------------------------------------------------
extern "C" void kernel_launch(void* const* d_in, const int* in_sizes, int n_in,
                              void* d_out, int out_size, void* d_ws, size_t ws_size,
                              hipStream_t stream)
{
    const float*     sup = (const float*)d_in[0];      // [S,T,D]
    const long long* lab = (const long long*)d_in[1];  // [S] int64
    const float*     tgt = (const float*)d_in[2];      // [Q,T,D]

    const int S = in_sizes[1];
    const int D = in_sizes[0] / (S * TT);
    const int Q = in_sizes[2] / (TT * D);
    const int C = out_size / Q;

    float* x2  = (float*)d_ws;               // Q*T
    float* y2  = x2 + (size_t)Q * TT;        // S*T
    float* cum = y2 + (size_t)S * TT;        // Q*S

    const int rows = Q * TT + S * TT;
    row_sqnorm_kernel<<<(rows + 7) / 8, 256, 0, stream>>>(tgt, sup, x2, y2,
                                                          Q * TT, S * TT, D);
    sdtw_pair_kernel<<<Q * S, 256, 0, stream>>>(tgt, sup, x2, y2, cum, S, D);
    aggregate_kernel<<<(Q + 127) / 128, 128, 0, stream>>>(cum, lab, (float*)d_out,
                                                          Q, S, C);
}
